// SingleGPUMoETorchFFN_63522566308131
// MI455X (gfx1250) — compile-verified, matrix operand order checked by reading, not run on hardware
//
#include <hip/hip_runtime.h>
#include <hip/hip_bf16.h>

typedef __attribute__((ext_vector_type(16))) __bf16 v16bf;
typedef __attribute__((ext_vector_type(8)))  float  v8f;

constexpr int kT = 2048;   // tokens
constexpr int kD = 1024;   // model dim
constexpr int kH = 3584;   // hidden dim
constexpr int kE = 8;      // experts
constexpr int BK = 32;     // K slice per WMMA step
constexpr int LDSK = 40;   // padded LDS row stride (ushorts) -> 80B, conflict-free b128 reads

#if defined(__has_builtin)
#  if __has_builtin(__builtin_amdgcn_cvt_pk_bf16_f32)
#    define HAVE_CVT_PK_BF16 1
#  endif
#endif

union FragBF {
  v16bf v;
  uint4 q[2];
};

__device__ inline unsigned short f32_to_bf16(float f) {
  unsigned u = __float_as_uint(f);
  u += 0x7FFFu + ((u >> 16) & 1u);   // round-to-nearest-even
  return (unsigned short)(u >> 16);
}
// Two f32 -> packed bf16x2 (elem0 in low half). RNE bias then one v_perm_b32
// to gather the two rounded high halves: bytes {ua[2],ua[3],ub[2],ub[3]}.
__device__ inline unsigned pack_bf16x2(float a, float b) {
#if HAVE_CVT_PK_BF16
  typedef __attribute__((ext_vector_type(2))) __bf16 v2bf;
  v2bf p = __builtin_amdgcn_cvt_pk_bf16_f32(a, b);
  return __builtin_bit_cast(unsigned, p);
#else
  unsigned ua = __float_as_uint(a);
  unsigned ub = __float_as_uint(b);
  ua += 0x7FFFu + ((ua >> 16) & 1u);
  ub += 0x7FFFu + ((ub >> 16) & 1u);
  return __builtin_amdgcn_perm(ub, ua, 0x07060302u);
#endif
}
__device__ inline uint4 pack8(float4 f0, float4 f1) {
  return make_uint4(pack_bf16x2(f0.x, f0.y), pack_bf16x2(f0.z, f0.w),
                    pack_bf16x2(f1.x, f1.y), pack_bf16x2(f1.z, f1.w));
}

// ---------------------------------------------------------------------------
// Kernel 0: zero output + routing counters
// ---------------------------------------------------------------------------
__global__ __launch_bounds__(512)
void moe_zero(float* __restrict__ out, int* __restrict__ counts) {
  const int gid = blockIdx.x * 512 + threadIdx.x;
  out[gid] = 0.f;
  if (gid < kE) counts[gid] = 0;
}

// ---------------------------------------------------------------------------
// Kernel 1: gating -> top-2 + softmax, append token to per-expert lists
// ---------------------------------------------------------------------------
__global__ __launch_bounds__(256)
void moe_gate(const float* __restrict__ x, const float* __restrict__ gw,
              int* __restrict__ counts, int* __restrict__ tok,
              float* __restrict__ wgt) {
  const int t = blockIdx.x;
  const int tid = threadIdx.x;
  __shared__ float part[kE * 256];
  __shared__ float sc[kE];

  float a[kE];
#pragma unroll
  for (int e = 0; e < kE; ++e) a[e] = 0.f;
  const float* xr = x + (size_t)t * kD;
#pragma unroll
  for (int j = 0; j < kD / 256; ++j) {
    const float xi = xr[tid + j * 256];
#pragma unroll
    for (int e = 0; e < kE; ++e) a[e] += xi * gw[e * kD + tid + j * 256];
  }
#pragma unroll
  for (int e = 0; e < kE; ++e) part[e * 256 + tid] = a[e];
  __syncthreads();
  if (tid < kE) {
    float s = 0.f;
    for (int j = 0; j < 256; ++j) s += part[tid * 256 + j];
    sc[tid] = s;
  }
  __syncthreads();
  if (tid == 0) {
    int i0 = 0;
    float s0 = sc[0];
    for (int e = 1; e < kE; ++e)
      if (sc[e] > s0) { s0 = sc[e]; i0 = e; }
    int i1 = -1;
    float s1 = -3.4e38f;
    for (int e = 0; e < kE; ++e)
      if (e != i0 && sc[e] > s1) { s1 = sc[e]; i1 = e; }
    // softmax over the two selected logits (s0 >= s1)
    const float w0 = 1.f / (1.f + __expf(s1 - s0));
    const float w1 = 1.f - w0;
    int p0 = atomicAdd(&counts[i0], 1);
    tok[i0 * kT + p0] = t;
    wgt[i0 * kT + p0] = w0;
    int p1 = atomicAdd(&counts[i1], 1);
    tok[i1 * kT + p1] = t;
    wgt[i1 * kT + p1] = w1;
  }
}

// ---------------------------------------------------------------------------
// Kernel 2: exclusive prefix over expert counts
// ---------------------------------------------------------------------------
__global__ void moe_offsets(const int* __restrict__ counts, int* __restrict__ offsets) {
  if (threadIdx.x == 0 && blockIdx.x == 0) {
    int s = 0;
    for (int e = 0; e < kE; ++e) { offsets[e] = s; s += counts[e]; }
    offsets[kE] = s;
  }
}

// ---------------------------------------------------------------------------
// Kernel 3: mid[row, h] = silu(x W1_e^T) * (x W3_e^T)  (bf16 out)
// grid (H/128, E), block 512 (16 waves, 4x4, 32x32 per wave)
// Double-buffered LDS (constant buffer indices), one barrier per K-step.
// ---------------------------------------------------------------------------
struct Stage1 {
  float4 xa0, xa1, w10, w11, w30, w31;
};

__device__ inline void g1_load(Stage1& s, const float* xr, const float* w1r,
                               const float* w3r, int k0, int c8) {
  if (xr) {
    s.xa0 = *(const float4*)(xr + k0 + c8);
    s.xa1 = *(const float4*)(xr + k0 + c8 + 4);
  } else {
    s.xa0 = make_float4(0.f, 0.f, 0.f, 0.f);
    s.xa1 = make_float4(0.f, 0.f, 0.f, 0.f);
  }
  s.w10 = *(const float4*)(w1r + k0 + c8);
  s.w11 = *(const float4*)(w1r + k0 + c8 + 4);
  s.w30 = *(const float4*)(w3r + k0 + c8);
  s.w31 = *(const float4*)(w3r + k0 + c8 + 4);
}

__device__ inline void g1_store(const Stage1& s, unsigned short* A,
                                unsigned short* B1, unsigned short* B3,
                                int ldsSt) {
  *(uint4*)(&A[ldsSt])  = pack8(s.xa0, s.xa1);
  *(uint4*)(&B1[ldsSt]) = pack8(s.w10, s.w11);
  *(uint4*)(&B3[ldsSt]) = pack8(s.w30, s.w31);
}

__device__ inline void g1_compute(const unsigned short* A, const unsigned short* B1,
                                  const unsigned short* B3, int rowBase, int colBase,
                                  int laneKb, v8f (&acc1)[2][2], v8f (&acc3)[2][2]) {
  FragBF a[2];
#pragma unroll
  for (int mt = 0; mt < 2; ++mt) {
    const int row = rowBase + mt * 16;
    a[mt].q[0] = *(const uint4*)(&A[row * LDSK + laneKb]);
    a[mt].q[1] = *(const uint4*)(&A[row * LDSK + laneKb + 16]);
  }
#pragma unroll
  for (int nt = 0; nt < 2; ++nt) {
    const int col = colBase + nt * 16;
    FragBF b1, b3;
    b1.q[0] = *(const uint4*)(&B1[col * LDSK + laneKb]);
    b1.q[1] = *(const uint4*)(&B1[col * LDSK + laneKb + 16]);
    b3.q[0] = *(const uint4*)(&B3[col * LDSK + laneKb]);
    b3.q[1] = *(const uint4*)(&B3[col * LDSK + laneKb + 16]);
#pragma unroll
    for (int mt = 0; mt < 2; ++mt) {
      acc1[mt][nt] = __builtin_amdgcn_wmma_f32_16x16x32_bf16(
          false, a[mt].v, false, b1.v, (short)0, acc1[mt][nt], false, false);
      acc3[mt][nt] = __builtin_amdgcn_wmma_f32_16x16x32_bf16(
          false, a[mt].v, false, b3.v, (short)0, acc3[mt][nt], false, false);
    }
  }
}

__global__ __launch_bounds__(512)
void moe_gemm1(const float* __restrict__ x, const float* __restrict__ W1,
               const float* __restrict__ W3, const int* __restrict__ counts,
               const int* __restrict__ offsets, const int* __restrict__ tok,
               unsigned short* __restrict__ mid) {
  const int e = blockIdx.y;
  const int hTile = blockIdx.x;
  const int n = counts[e];
  if (n == 0) return;
  const int off = offsets[e];
  const int tid = threadIdx.x;
  const int lane = tid & 31;
  const int wid = tid >> 5;
  const int waveM = wid >> 2;
  const int waveN = wid & 3;

  __shared__ __align__(16) unsigned short As[2][128 * LDSK];
  __shared__ __align__(16) unsigned short B1s[2][128 * LDSK];
  __shared__ __align__(16) unsigned short B3s[2][128 * LDSK];

  const int r  = tid >> 2;        // 0..127 tile row handled by this thread (coop loads)
  const int c8 = (tid & 3) * 8;   // 0,8,16,24

  const float* w1r = W1 + ((size_t)e * kH + (size_t)(hTile * 128 + r)) * kD;
  const float* w3r = W3 + ((size_t)e * kH + (size_t)(hTile * 128 + r)) * kD;

  const int laneRow = lane & 15;
  const int laneKb  = (lane >> 4) * 8;
  const int ldsSt   = r * LDSK + c8;
  const int rowBase = waveM * 32 + laneRow;
  const int colBase = waveN * 32 + laneRow;

  constexpr int KITERS = kD / BK;   // 32 (even)

  for (int m0 = 0; m0 < n; m0 += 128) {
    v8f acc1[2][2] = {};
    v8f acc3[2][2] = {};
    const int trow = m0 + r;
    const float* xr = nullptr;
    if (trow < n) xr = x + (size_t)tok[e * kT + trow] * kD;

    // ---- prologue: stage K-slice 0 into buffer 0 ----
    Stage1 st;
    g1_load(st, xr, w1r, w3r, 0, c8);
    g1_store(st, As[0], B1s[0], B3s[0], ldsSt);
    __syncthreads();

    for (int ki = 0; ki < KITERS; ki += 2) {
      // phase A: compute buf0, stage slice ki+1 -> buf1
      g1_load(st, xr, w1r, w3r, (ki + 1) * BK, c8);
      g1_compute(As[0], B1s[0], B3s[0], rowBase, colBase, laneKb, acc1, acc3);
      g1_store(st, As[1], B1s[1], B3s[1], ldsSt);
      __syncthreads();
      // phase B: compute buf1, stage slice ki+2 -> buf0 (if any)
      const bool more = (ki + 2) < KITERS;
      if (more) g1_load(st, xr, w1r, w3r, (ki + 2) * BK, c8);
      g1_compute(As[1], B1s[1], B3s[1], rowBase, colBase, laneKb, acc1, acc3);
      if (more) g1_store(st, As[0], B1s[0], B3s[0], ldsSt);
      __syncthreads();
    }

    // ---- fused SwiGLU epilogue -> bf16 mid ----
#pragma unroll
    for (int mt = 0; mt < 2; ++mt) {
      const int rB = m0 + waveM * 32 + mt * 16 + (lane >> 4) * 8;
#pragma unroll
      for (int nt = 0; nt < 2; ++nt) {
        const int col = hTile * 128 + waveN * 32 + nt * 16 + laneRow;
#pragma unroll
        for (int rr = 0; rr < 8; ++rr) {
          const int row = rB + rr;
          if (row < n) {
            const float h1 = acc1[mt][nt][rr];
            const float val = (h1 / (1.f + __expf(-h1))) * acc3[mt][nt][rr];
            mid[(size_t)(off + row) * kH + col] = f32_to_bf16(val);
          }
        }
      }
    }
  }
}

// ---------------------------------------------------------------------------
// Kernel 4: out[t, d] += w * (mid W2_e^T)   grid (D/128, E), block 512
// Double-buffered LDS (constant buffer indices), one barrier per K-step.
// ---------------------------------------------------------------------------
struct Stage2 {
  uint4  ma;          // mid A slice (already bf16)
  float4 w20, w21;    // W2 slice (f32)
};

__device__ inline void g2_load(Stage2& s, const unsigned short* midr,
                               const float* w2r, int k0, int c8) {
  s.ma = midr ? *(const uint4*)(midr + k0 + c8) : make_uint4(0u, 0u, 0u, 0u);
  s.w20 = *(const float4*)(w2r + k0 + c8);
  s.w21 = *(const float4*)(w2r + k0 + c8 + 4);
}

__device__ inline void g2_store(const Stage2& s, unsigned short* A,
                                unsigned short* B, int ldsSt) {
  *(uint4*)(&A[ldsSt]) = s.ma;
  *(uint4*)(&B[ldsSt]) = pack8(s.w20, s.w21);
}

__device__ inline void g2_compute(const unsigned short* A, const unsigned short* B,
                                  int rowBase, int colBase, int laneKb,
                                  v8f (&acc)[2][2]) {
  FragBF a[2];
#pragma unroll
  for (int mt = 0; mt < 2; ++mt) {
    const int row = rowBase + mt * 16;
    a[mt].q[0] = *(const uint4*)(&A[row * LDSK + laneKb]);
    a[mt].q[1] = *(const uint4*)(&A[row * LDSK + laneKb + 16]);
  }
#pragma unroll
  for (int nt = 0; nt < 2; ++nt) {
    const int col = colBase + nt * 16;
    FragBF b;
    b.q[0] = *(const uint4*)(&B[col * LDSK + laneKb]);
    b.q[1] = *(const uint4*)(&B[col * LDSK + laneKb + 16]);
#pragma unroll
    for (int mt = 0; mt < 2; ++mt) {
      acc[mt][nt] = __builtin_amdgcn_wmma_f32_16x16x32_bf16(
          false, a[mt].v, false, b.v, (short)0, acc[mt][nt], false, false);
    }
  }
}

__global__ __launch_bounds__(512)
void moe_gemm2(const unsigned short* __restrict__ mid, const float* __restrict__ W2,
               const int* __restrict__ counts, const int* __restrict__ offsets,
               const int* __restrict__ tok, const float* __restrict__ wgt,
               float* __restrict__ out) {
  const int e = blockIdx.y;
  const int dTile = blockIdx.x;
  const int n = counts[e];
  if (n == 0) return;
  const int off = offsets[e];
  const int tid = threadIdx.x;
  const int lane = tid & 31;
  const int wid = tid >> 5;
  const int waveM = wid >> 2;
  const int waveN = wid & 3;

  __shared__ __align__(16) unsigned short As[2][128 * LDSK];
  __shared__ __align__(16) unsigned short Bs[2][128 * LDSK];

  const int r  = tid >> 2;
  const int c8 = (tid & 3) * 8;

  const float* w2r = W2 + ((size_t)e * kD + (size_t)(dTile * 128 + r)) * kH;

  const int laneRow = lane & 15;
  const int laneKb  = (lane >> 4) * 8;
  const int ldsSt   = r * LDSK + c8;
  const int rowBase = waveM * 32 + laneRow;
  const int colBase = waveN * 32 + laneRow;

  constexpr int KITERS = kH / BK;   // 112 (even)

  for (int m0 = 0; m0 < n; m0 += 128) {
    v8f acc[2][2] = {};
    const int trow = m0 + r;
    const unsigned short* midr =
        (trow < n) ? (mid + (size_t)(off + trow) * kH) : nullptr;

    // ---- prologue ----
    Stage2 st;
    g2_load(st, midr, w2r, 0, c8);
    g2_store(st, As[0], Bs[0], ldsSt);
    __syncthreads();

    for (int ki = 0; ki < KITERS; ki += 2) {
      g2_load(st, midr, w2r, (ki + 1) * BK, c8);
      g2_compute(As[0], Bs[0], rowBase, colBase, laneKb, acc);
      g2_store(st, As[1], Bs[1], ldsSt);
      __syncthreads();

      const bool more = (ki + 2) < KITERS;
      if (more) g2_load(st, midr, w2r, (ki + 2) * BK, c8);
      g2_compute(As[1], Bs[1], rowBase, colBase, laneKb, acc);
      if (more) g2_store(st, As[0], Bs[0], ldsSt);
      __syncthreads();
    }

    // ---- weighted scatter-add into output ----
#pragma unroll
    for (int mt = 0; mt < 2; ++mt) {
      const int rB = m0 + waveM * 32 + mt * 16 + (lane >> 4) * 8;
#pragma unroll
      for (int rr = 0; rr < 8; ++rr) {
        const int row = rB + rr;
        if (row < n) {
          const int t = tok[e * kT + row];
          const float w = wgt[e * kT + row];
#pragma unroll
          for (int nt = 0; nt < 2; ++nt) {
            const int col = dTile * 128 + waveN * 32 + nt * 16 + laneRow;
            __hip_atomic_fetch_add(&out[(size_t)t * kD + col], w * acc[mt][nt][rr],
                                   __ATOMIC_RELAXED, __HIP_MEMORY_SCOPE_AGENT);
          }
        }
      }
    }
  }
}

// ---------------------------------------------------------------------------
extern "C" void kernel_launch(void* const* d_in, const int* in_sizes, int n_in,
                              void* d_out, int out_size, void* d_ws, size_t ws_size,
                              hipStream_t stream) {
  const float* x  = (const float*)d_in[0];
  const float* gw = (const float*)d_in[1];
  const float* W1 = (const float*)d_in[2];
  const float* W2 = (const float*)d_in[3];
  const float* W3 = (const float*)d_in[4];
  float* out = (float*)d_out;

  // workspace layout
  char* ws = (char*)d_ws;
  int*   counts  = (int*)ws;                               // 8 ints
  int*   offsets = (int*)(ws + 64);                        // 9 ints
  int*   tok     = (int*)(ws + 128);                       // kE*kT ints   (64 KB)
  float* wgt     = (float*)(ws + 128 + kE * kT * 4);       // kE*kT floats (64 KB)
  unsigned short* mid = (unsigned short*)(ws + (1 << 20)); // 2*kT*kH bf16 (~29.4 MB)

  moe_zero<<<(kT * kD) / 512, 512, 0, stream>>>(out, counts);
  moe_gate<<<kT, 256, 0, stream>>>(x, gw, counts, tok, wgt);
  moe_offsets<<<1, 32, 0, stream>>>(counts, offsets);
  moe_gemm1<<<dim3(kH / 128, kE), 512, 0, stream>>>(x, W1, W3, counts, offsets, tok, mid);
  moe_gemm2<<<dim3(kD / 128, kE), 512, 0, stream>>>(mid, W2, counts, offsets, tok, wgt, out);
}
